// DecoderLayer_83691732730276
// MI455X (gfx1250) — matrix operand order repaired and checked by reading, not compiled, not run on hardware
//
#include <hip/hip_runtime.h>
#include <hip/hip_bf16.h>
#include <cstdint>
#include <cstddef>

// ---------- types ----------
typedef __bf16 bf16_t;
typedef __attribute__((ext_vector_type(16))) __bf16 v16bf;
typedef __attribute__((ext_vector_type(8)))  __bf16 v8bf;
typedef __attribute__((ext_vector_type(8)))  float  v8f;

#define NEGC (-1.0e9f)

// ---------- problem constants ----------
#define CB    8
#define CLD   1024
#define CLE   1024
#define CD    1024
#define CH_   16
#define CDK   64
#define CDFF  4096
#define CM    (CB * CLD)   // 8192 token rows

__device__ __forceinline__ v8f wmma_bf16(v16bf a, v16bf b, v8f c) {
  // D = A(16x32 bf16) * B(32x16 bf16) + C(16x16 f32)
  return __builtin_amdgcn_wmma_f32_16x16x32_bf16(false, a, false, b, (short)0, c,
                                                 false, false);
}

// ---------- CDNA5 async global->LDS copy (ASYNCcnt path) ----------
__device__ __forceinline__ void async_cp16(void* lds_ptr, const void* gptr) {
  unsigned int loff = (unsigned int)(uintptr_t)lds_ptr;            // LDS byte offset
  unsigned long long ga = (unsigned long long)(uintptr_t)gptr;     // global address
  asm volatile("global_load_async_to_lds_b128 %0, %1, off"
               :: "v"(loff), "v"(ga)
               : "memory");
}

__device__ __forceinline__ void wait_async0() {
#if __has_builtin(__builtin_amdgcn_s_wait_asynccnt)
  __builtin_amdgcn_s_wait_asynccnt(0);
#else
  asm volatile("s_wait_asynccnt 0" ::: "memory");
#endif
}

// A fragment: lane holds row m; j=0..15 -> K = k0 + 16*hi + j (contiguous).
__device__ __forceinline__ v16bf load_a_frag(const bf16_t* base, int ld, int m,
                                             int k0, int hi) {
  const bf16_t* p = base + (size_t)m * ld + k0 + hi * 16;
  v8bf lo = *(const v8bf*)p;
  v8bf hh = *(const v8bf*)(p + 8);
  v16bf r;
#pragma unroll
  for (int i = 0; i < 8; ++i) { r[i] = lo[i]; r[8 + i] = hh[i]; }
  return r;
}

// Same A-fragment pattern but sourced from f32 LDS (softmax P), cvt to bf16.
__device__ __forceinline__ v16bf load_p_frag(const float* s, int ld, int m,
                                             int k0, int hi) {
  const float* p = s + (size_t)m * ld + k0 + hi * 16;
  v16bf r;
#pragma unroll
  for (int i = 0; i < 8; ++i) { r[i] = (bf16_t)p[i]; r[8 + i] = (bf16_t)p[8 + i]; }
  return r;
}

// ---------- elementwise f32 -> bf16 ----------
__global__ void cvt_bf16_kernel(const float* __restrict__ in, bf16_t* __restrict__ out,
                                size_t n) {
  size_t i = (size_t)blockIdx.x * blockDim.x + threadIdx.x;
  size_t step = (size_t)gridDim.x * blockDim.x;
  for (; i < n; i += step) out[i] = (bf16_t)in[i];
}

// ---------- transpose + convert: out[z][c][r] = in[z][r][c] ----------
__global__ void transpose_cvt_kernel(const float* __restrict__ in,
                                     bf16_t* __restrict__ out, int R, int C) {
  size_t slice = (size_t)R * C;
  const float* ip = in + (size_t)blockIdx.z * slice;
  bf16_t* op = out + (size_t)blockIdx.z * slice;
  size_t i = (size_t)blockIdx.x * blockDim.x + threadIdx.x;
  size_t step = (size_t)gridDim.x * blockDim.x;
  for (; i < slice; i += step) {
    int r = (int)(i / C), c = (int)(i % C);
    op[(size_t)c * R + r] = (bf16_t)ip[i];
  }
}

// ---------- block-tiled WMMA GEMM with double-buffered async LDS staging ----------
// C[m,n] = act( (sum_k A[m,k] * Bt[n,k]) + bias[n] ) * alpha
// A: [M,K] bf16 row-major.  Bt: [Z][N,K] bf16 (B transposed).  bias: [Z][N] f32.
// Cb (bf16): transC ? [Z][N,M] : [Z][M,N].  Cf (f32): [M,ldcf].
// Block = 8 waves; block tile = BM x BN; each wave computes 32 x 64.
// WN=1: BM=256, BN=64 (QKV projections).  WN=2: BM=128, BN=128 (FFN).
template <int WN>
__global__ __launch_bounds__(256) void gemm_tile_kernel(
    const bf16_t* __restrict__ A, const bf16_t* __restrict__ Bt,
    const float* __restrict__ bias, float alpha, int relu, int transC,
    bf16_t* __restrict__ Cb, float* __restrict__ Cf,
    int M, int N, int K, int ldcf) {
  constexpr int WM = 8 / WN;
  constexpr int BM = 32 * WM;
  constexpr int BN = 64 * WN;
  constexpr int LDT = 40;   // padded LDS row stride (elems) for a 32-wide k panel

  __shared__ __align__(16) bf16_t sA[2][BM][LDT];
  __shared__ __align__(16) bf16_t sB[2][BN][LDT];

  int tid = threadIdx.x;
  int lane = tid & 31, wave = tid >> 5;
  int hi = lane >> 4, ln = lane & 15;
  int wm = wave % WM, wn = wave / WM;

  int n0 = blockIdx.x * BN, m0 = blockIdx.y * BM, z = blockIdx.z;
  const bf16_t* Az = A + (size_t)m0 * K;
  const bf16_t* Bz = Bt + (size_t)z * N * K + (size_t)n0 * K;
  const float* bz = bias ? bias + (size_t)z * N : nullptr;
  const int NK = K / 32;

  auto stage = [&](int buf, int ks) {
    int kof = ks * 32;
    for (int i = tid; i < BM * 2; i += 256) {       // A tile: BM x 32
      int r = i >> 1, cc = i & 1;
      async_cp16(&sA[buf][r][cc * 16], Az + (size_t)r * K + kof + cc * 16);
    }
    for (int i = tid; i < BN * 2; i += 256) {       // B tile: BN x 32
      int r = i >> 1, cc = i & 1;
      async_cp16(&sB[buf][r][cc * 16], Bz + (size_t)r * K + kof + cc * 16);
    }
  };

  v8f acc[2][4] = {};

  stage(0, 0);
  wait_async0();
  __syncthreads();

  for (int ks = 0; ks < NK; ++ks) {
    int cur = ks & 1;
    if (ks + 1 < NK) stage(cur ^ 1, ks + 1);   // overlap DMA with compute

    v16bf af[2];
#pragma unroll
    for (int tm = 0; tm < 2; ++tm)
      af[tm] = load_a_frag(&sA[cur][0][0], LDT, wm * 32 + tm * 16 + ln, 0, hi);
#pragma unroll
    for (int tn = 0; tn < 4; ++tn) {
      v16bf bfr = *(const v16bf*)(&sB[cur][wn * 64 + tn * 16 + ln][hi * 16]);
#pragma unroll
      for (int tm = 0; tm < 2; ++tm)
        acc[tm][tn] = wmma_bf16(af[tm], bfr, acc[tm][tn]);
    }

    wait_async0();
    __syncthreads();
  }

#pragma unroll
  for (int tm = 0; tm < 2; ++tm)
#pragma unroll
    for (int tn = 0; tn < 4; ++tn)
#pragma unroll
      for (int r = 0; r < 8; ++r) {
        int mm = m0 + wm * 32 + tm * 16 + r + 8 * hi;
        int nn = n0 + wn * 64 + tn * 16 + ln;
        float v = acc[tm][tn][r];
        if (bz) v += bz[nn];
        v *= alpha;
        if (relu) v = fmaxf(v, 0.0f);
        if (Cb) {
          size_t idx = transC ? ((size_t)z * N * M + (size_t)nn * M + mm)
                              : ((size_t)z * M * N + (size_t)mm * N + nn);
          Cb[idx] = (bf16_t)v;
        }
        if (Cf) Cf[(size_t)mm * ldcf + nn] = v;
      }
}

// ---------- flash attention (online softmax), one (b,h,64-row q tile) per block ----
#define KVCH 128
__global__ __launch_bounds__(256) void attn_wmma_kernel(
    const bf16_t* __restrict__ Qb,  // [H][B*Lq][64]  (pre-scaled by 1/sqrt(DK))
    const bf16_t* __restrict__ Kb,  // [H][B*Lk][64]
    const bf16_t* __restrict__ Vt,  // [H][64][B*Lk]
    float* __restrict__ Out,        // [B][Lq][H*64]
    int Bn, int Hn, int Lq, int Lk, int masked) {
  __shared__ __align__(16) bf16_t sQ[64][72];
  __shared__ __align__(32) float sS[64][KVCH + 8];
  __shared__ float rmax[64], rsum[64], rscale[64];

  int tid = threadIdx.x;
  int lane = tid & 31, wave = tid >> 5;
  int hi = lane >> 4, ln = lane & 15;
  int qt = blockIdx.x, h = blockIdx.y, b = blockIdx.z;

  size_t Mq = (size_t)Bn * Lq, Mk = (size_t)Bn * Lk;
  const bf16_t* Qh = Qb + ((size_t)h * Mq + (size_t)b * Lq + (size_t)qt * 64) * 64;
  const bf16_t* Kh = Kb + ((size_t)h * Mk + (size_t)b * Lk) * 64;
  const bf16_t* Vh = Vt + (size_t)h * 64 * Mk + (size_t)b * Lk;

  for (int i = tid; i < 64 * 64; i += 256) {
    int r = i >> 6, cc = i & 63;
    sQ[r][cc] = Qh[(size_t)r * 64 + cc];
  }
  if (tid < 64) { rmax[tid] = -3.0e38f; rsum[tid] = 0.0f; }
  __syncthreads();

  int om = wave & 3;
  int on0 = wave >> 2;
  int on1 = on0 + 2;
  v8f o0 = {}, o1 = {};

  int kvend = masked ? ((qt * 64 + 64 < Lk) ? qt * 64 + 64 : Lk) : Lk;
  for (int kc = 0; kc < kvend; kc += KVCH) {
    int cw = kvend - kc; if (cw > KVCH) cw = KVCH;   // multiple of 32 here

    // ---- S = Q * K^T tiles (64 x cw) ----
    for (int t = wave; t < 4 * (KVCH / 16); t += 8) {
      int tm = t & 3, tn = t >> 2;
      if (tn * 16 < cw) {
        v8f s = {};
#pragma unroll
        for (int k0 = 0; k0 < CDK; k0 += 32) {
          v16bf a = load_a_frag(&sQ[0][0], 72, tm * 16 + ln, k0, hi);
          v16bf bb = *(const v16bf*)(Kh + (size_t)(kc + tn * 16 + ln) * 64 + k0 + hi * 16);
          s = wmma_bf16(a, bb, s);
        }
#pragma unroll
        for (int r = 0; r < 8; ++r) {
          int qr = tm * 16 + r + 8 * hi;
          int kcol = tn * 16 + ln;
          float v = s[r];
          if (masked && (kc + kcol) > (qt * 64 + qr)) v += NEGC;
          sS[qr][kcol] = v;
        }
      }
    }
    __syncthreads();

    // ---- online softmax update (one thread per q row) ----
    if (tid < 64) {
      float mx = rmax[tid];
      float cm = -3.0e38f;
      for (int j = 0; j < cw; ++j) cm = fmaxf(cm, sS[tid][j]);
      float nm = fmaxf(mx, cm);
      float sc = __expf(mx - nm);
      float ps = 0.0f;
      for (int j = 0; j < cw; ++j) {
        float p = __expf(sS[tid][j] - nm);
        sS[tid][j] = p;
        ps += p;
      }
      rsum[tid] = rsum[tid] * sc + ps;
      rscale[tid] = sc;
      rmax[tid] = nm;
    }
    __syncthreads();

    // ---- O = O*scale + P * V ----
#pragma unroll
    for (int r = 0; r < 8; ++r) {
      float sc = rscale[om * 16 + r + 8 * hi];
      o0[r] *= sc;
      o1[r] *= sc;
    }
    for (int k0 = 0; k0 < cw; k0 += 32) {
      v16bf a = load_p_frag(&sS[0][0], KVCH + 8, om * 16 + ln, k0, hi);
      v16bf b0 = *(const v16bf*)(Vh + (size_t)(on0 * 16 + ln) * Mk + kc + k0 + hi * 16);
      o0 = wmma_bf16(a, b0, o0);
      v16bf b1 = *(const v16bf*)(Vh + (size_t)(on1 * 16 + ln) * Mk + kc + k0 + hi * 16);
      o1 = wmma_bf16(a, b1, o1);
    }
    __syncthreads();
  }

#pragma unroll
  for (int r = 0; r < 8; ++r) {
    int qr = om * 16 + r + 8 * hi;
    float inv = 1.0f / rsum[qr];
    size_t row = (size_t)b * Lq + (size_t)qt * 64 + qr;
    Out[row * (size_t)(Hn * 64) + (size_t)h * 64 + on0 * 16 + ln] = o0[r] * inv;
    Out[row * (size_t)(Hn * 64) + (size_t)h * 64 + on1 * 16 + ln] = o1[r] * inv;
  }
}

// ---------- residual add + LayerNorm (one row per block) ----------
__global__ __launch_bounds__(256) void add_ln_kernel(
    const float* __restrict__ X, const float* __restrict__ R,
    const float* __restrict__ g, const float* __restrict__ be,
    float* __restrict__ Yf, bf16_t* __restrict__ Yb, int Dd) {
  __shared__ float red[256];
  int row = blockIdx.x, tid = threadIdx.x;
  const float* x = X + (size_t)row * Dd;
  const float* rr = R + (size_t)row * Dd;

  float s = 0.0f;
  for (int i = tid; i < Dd; i += 256) s += x[i] + rr[i];
  red[tid] = s; __syncthreads();
  for (int w = 128; w > 0; w >>= 1) { if (tid < w) red[tid] += red[tid + w]; __syncthreads(); }
  float mean = red[0] / Dd;
  __syncthreads();

  float v = 0.0f;
  for (int i = tid; i < Dd; i += 256) { float d = x[i] + rr[i] - mean; v += d * d; }
  red[tid] = v; __syncthreads();
  for (int w = 128; w > 0; w >>= 1) { if (tid < w) red[tid] += red[tid + w]; __syncthreads(); }
  float inv = rsqrtf(red[0] / Dd + 1e-5f);

  for (int i = tid; i < Dd; i += 256) {
    float y = (x[i] + rr[i] - mean) * inv * g[i] + be[i];
    if (Yf) Yf[(size_t)row * Dd + i] = y;
    if (Yb) Yb[(size_t)row * Dd + i] = (bf16_t)y;
  }
}

// =====================================================================
extern "C" void kernel_launch(void* const* d_in, const int* in_sizes, int n_in,
                              void* d_out, int out_size, void* d_ws, size_t ws_size,
                              hipStream_t stream) {
  const float* dec   = (const float*)d_in[0];
  const float* enc   = (const float*)d_in[1];
  const float* m_wq  = (const float*)d_in[2];
  const float* m_bq  = (const float*)d_in[3];
  const float* m_wk  = (const float*)d_in[4];
  const float* m_bk  = (const float*)d_in[5];
  const float* m_wv  = (const float*)d_in[6];
  const float* m_bv  = (const float*)d_in[7];
  const float* c_wq  = (const float*)d_in[8];
  const float* c_bq  = (const float*)d_in[9];
  const float* c_wk  = (const float*)d_in[10];
  const float* c_bk  = (const float*)d_in[11];
  const float* c_wv  = (const float*)d_in[12];
  const float* c_bv  = (const float*)d_in[13];
  const float* ff_w1 = (const float*)d_in[14];
  const float* ff_b1 = (const float*)d_in[15];
  const float* ff_w2 = (const float*)d_in[16];
  const float* ff_b2 = (const float*)d_in[17];
  const float* ln1_g = (const float*)d_in[18];
  const float* ln1_b = (const float*)d_in[19];
  const float* ln2_g = (const float*)d_in[20];
  const float* ln2_b = (const float*)d_in[21];
  float* out = (float*)d_out;

  // ---- workspace carve-up ----
  char* ws = (char*)d_ws;
  size_t off = 0;
  auto take = [&](size_t bytes) -> char* {
    char* p = ws + off;
    off += (bytes + 255) & ~(size_t)255;
    return p;
  };
  const size_t NTOK = (size_t)CM;               // 8192 rows
  bf16_t* decb = (bf16_t*)take(NTOK * CD * 2);
  bf16_t* encb = (bf16_t*)take(NTOK * CD * 2);
  bf16_t* mqt  = (bf16_t*)take((size_t)CH_ * CD * CDK * 2);
  bf16_t* mkt  = (bf16_t*)take((size_t)CH_ * CD * CDK * 2);
  bf16_t* mvt  = (bf16_t*)take((size_t)CH_ * CD * CDK * 2);
  bf16_t* cqt  = (bf16_t*)take((size_t)CH_ * CD * CDK * 2);
  bf16_t* ckt  = (bf16_t*)take((size_t)CH_ * CD * CDK * 2);
  bf16_t* cvt_ = (bf16_t*)take((size_t)CH_ * CD * CDK * 2);
  bf16_t* w1t  = (bf16_t*)take((size_t)CD * CDFF * 2);
  bf16_t* w2t  = (bf16_t*)take((size_t)CDFF * CD * 2);
  bf16_t* Qb   = (bf16_t*)take((size_t)CH_ * NTOK * CDK * 2);
  bf16_t* Kb   = (bf16_t*)take((size_t)CH_ * NTOK * CDK * 2);
  bf16_t* Vtb  = (bf16_t*)take((size_t)CH_ * CDK * NTOK * 2);
  float*  attn = (float*)take(NTOK * CD * 4);   // reused: self-attn, cross-attn, ffn2
  float*  x1f  = (float*)take(NTOK * CD * 4);
  bf16_t* x1b  = (bf16_t*)take(NTOK * CD * 2);
  float*  x2f  = (float*)take(NTOK * CD * 4);
  bf16_t* x2b  = (bf16_t*)take(NTOK * CD * 2);
  bf16_t* hbuf = (bf16_t*)take(NTOK * CDFF * 2);
  (void)ws_size;

  const float qscale = 0.125f;   // 1/sqrt(64)

  // 1) activations -> bf16
  cvt_bf16_kernel<<<4096, 256, 0, stream>>>(dec, decb, NTOK * CD);
  cvt_bf16_kernel<<<4096, 256, 0, stream>>>(enc, encb, NTOK * CD);

  // 2) weights -> transposed bf16 (Bt[n][k])
  transpose_cvt_kernel<<<dim3(256, 1, CH_), 256, 0, stream>>>(m_wq, mqt, CD, CDK);
  transpose_cvt_kernel<<<dim3(256, 1, CH_), 256, 0, stream>>>(m_wk, mkt, CD, CDK);
  transpose_cvt_kernel<<<dim3(256, 1, CH_), 256, 0, stream>>>(m_wv, mvt, CD, CDK);
  transpose_cvt_kernel<<<dim3(256, 1, CH_), 256, 0, stream>>>(c_wq, cqt, CD, CDK);
  transpose_cvt_kernel<<<dim3(256, 1, CH_), 256, 0, stream>>>(c_wk, ckt, CD, CDK);
  transpose_cvt_kernel<<<dim3(256, 1, CH_), 256, 0, stream>>>(c_wv, cvt_, CD, CDK);
  transpose_cvt_kernel<<<dim3(4096, 1, 1), 256, 0, stream>>>(ff_w1, w1t, CD, CDFF);
  transpose_cvt_kernel<<<dim3(4096, 1, 1), 256, 0, stream>>>(ff_w2, w2t, CDFF, CD);

  // 3) self-attention QKV projections (block tile 256x64, 16 heads on grid.z)
  dim3 gQKV(CDK / 64, CM / 256, CH_);
  gemm_tile_kernel<1><<<gQKV, 256, 0, stream>>>(decb, mqt, m_bq, qscale, 0, 0,
                                                Qb, nullptr, CM, CDK, CD, 0);
  gemm_tile_kernel<1><<<gQKV, 256, 0, stream>>>(decb, mkt, m_bk, 1.0f, 0, 0,
                                                Kb, nullptr, CM, CDK, CD, 0);
  gemm_tile_kernel<1><<<gQKV, 256, 0, stream>>>(decb, mvt, m_bv, 1.0f, 0, 1,
                                                Vtb, nullptr, CM, CDK, CD, 0);

  // 4) masked self-attention
  attn_wmma_kernel<<<dim3(CLD / 64, CH_, CB), 256, 0, stream>>>(
      Qb, Kb, Vtb, attn, CB, CH_, CLD, CLD, 1);

  // 5) x1 = LN1(dec + attn)
  add_ln_kernel<<<CM, 256, 0, stream>>>(dec, attn, ln1_g, ln1_b, x1f, x1b, CD);

  // 6) cross-attention QKV (Q from x1, K/V from encoder)
  gemm_tile_kernel<1><<<gQKV, 256, 0, stream>>>(x1b, cqt, c_bq, qscale, 0, 0,
                                                Qb, nullptr, CM, CDK, CD, 0);
  gemm_tile_kernel<1><<<gQKV, 256, 0, stream>>>(encb, ckt, c_bk, 1.0f, 0, 0,
                                                Kb, nullptr, CM, CDK, CD, 0);
  gemm_tile_kernel<1><<<gQKV, 256, 0, stream>>>(encb, cvt_, c_bv, 1.0f, 0, 1,
                                                Vtb, nullptr, CM, CDK, CD, 0);

  // 7) cross-attention (no mask)
  attn_wmma_kernel<<<dim3(CLD / 64, CH_, CB), 256, 0, stream>>>(
      Qb, Kb, Vtb, attn, CB, CH_, CLD, CLE, 0);

  // 8) x2 = LN2(x1 + attn)
  add_ln_kernel<<<CM, 256, 0, stream>>>(x1f, attn, ln2_g, ln2_b, x2f, x2b, CD);

  // 9) FFN: h = relu(x2 @ W1 + b1)   (block tile 128x128)
  gemm_tile_kernel<2><<<dim3(CDFF / 128, CM / 128, 1), 256, 0, stream>>>(
      x2b, w1t, ff_b1, 1.0f, 1, 0, hbuf, nullptr, CM, CDFF, CD, 0);

  // 10) f = h @ W2 + b2   (f32 into `attn` buffer)
  gemm_tile_kernel<2><<<dim3(CD / 128, CM / 128, 1), 256, 0, stream>>>(
      hbuf, w2t, ff_b2, 1.0f, 0, 0, nullptr, attn, CM, CD, CDFF, CD);

  // 11) out = LN2(x2 + f)
  add_ln_kernel<<<CM, 256, 0, stream>>>(x2f, attn, ln2_g, ln2_b, out, nullptr, CD);
}
